// DenseIpaScore_84731114816265
// MI455X (gfx1250) — matrix operand order repaired and checked by reading, not compiled, not run on hardware
//
#include <hip/hip_runtime.h>
#include <hip/hip_bf16.h>
#include <math.h>

// ---------------------------------------------------------------------------
// MI455X (gfx1250) implementation. All dense linears run through a wave32
// WMMA GEMM (v_wmma_f32_16x16x32_f16): activations cast f32->f16 (K padded to
// 32), weights packed transposed [dout,din] in f16 so every lane's A/B
// fragment is two contiguous 16B loads (ISA 7.12.2 layouts). Each wave now
// computes a 32x32 register tile (2x2 fragments, 4 accumulators) so 8 b128
// loads feed 4 WMMAs (16 FLOP/byte); 8 waves/block produce a 128x64 f32 tile.
// Edge-transition GEMMs (the FLOP hotspot, ~135 GFLOP) are chunked 8192 pairs
// at a time to bound workspace.
// ---------------------------------------------------------------------------

typedef _Float16 half_t;
typedef _Float16 v8h  __attribute__((ext_vector_type(8)));
typedef _Float16 v16h __attribute__((ext_vector_type(16)));
typedef float    v8f  __attribute__((ext_vector_type(8)));

#define N_    256
#define H_    8
#define PQ_   8
#define PV_   12
#define CZ_   128
#define INFB  100000.0f
#define SCALE_ 0.1f

static inline int pad16i(int x){ return (x+15)&~15; }
static inline int pad32i(int x){ return (x+31)&~31; }

// ------------------------------ device helpers -----------------------------
__device__ __forceinline__ void quatR(const float* q, float* R){
  float w=q[0],x=q[1],y=q[2],z=q[3];
  R[0]=1.f-2.f*(y*y+z*z); R[1]=2.f*(x*y-w*z);     R[2]=2.f*(x*z+w*y);
  R[3]=2.f*(x*y+w*z);     R[4]=1.f-2.f*(x*x+z*z); R[5]=2.f*(y*z-w*x);
  R[6]=2.f*(x*z-w*y);     R[7]=2.f*(y*z+w*x);     R[8]=1.f-2.f*(x*x+y*y);
}
// A fragment: lane holds M-row (r16), K slices [kh*8, kh*8+8) and [16+kh*8, ..)
__device__ __forceinline__ v16h ldA(const half_t* __restrict__ R0, int k, int kh){
  v8h lo = *(const v8h*)(R0 + k + kh*8);
  v8h hi = *(const v8h*)(R0 + k + 16 + kh*8);
  return __builtin_shufflevector(lo,hi,0,1,2,3,4,5,6,7,8,9,10,11,12,13,14,15);
}
// B fragment: lane holds N-row (r16), K slices [kh*8, kh*8+8) and [16+kh*8, ..)
__device__ __forceinline__ v16h ldB(const half_t* __restrict__ R0, int k, int kh){
  v8h lo = *(const v8h*)(R0 + k + kh*8);
  v8h hi = *(const v8h*)(R0 + k + 16 + kh*8);
  return __builtin_shufflevector(lo,hi,0,1,2,3,4,5,6,7,8,9,10,11,12,13,14,15);
}
#define WMMA_(A,B,C) __builtin_amdgcn_wmma_f32_16x16x32_f16(false, A, false, B, \
                                                            (short)0, C, false, false)

// ------------------------------- GEMM (WMMA) -------------------------------
// C[M,Np](f32,ldc) = alpha * A(f16 [M,lda]) x B(f16 [Np,ldb])^T (+bias)(+res),
// optional ReLU. K multiple of 32, M multiple of 32 (all call sites: 128).
// Columns >= Nstore are not written. Wave tile 32x32, block tile 128x64.
__global__ __launch_bounds__(256)
void k_gemm(const half_t* __restrict__ A, int lda,
            const half_t* __restrict__ B, int ldb,
            float* __restrict__ C, int ldc,
            const float* __restrict__ bias,
            const float* __restrict__ res, int ldres,
            int M, int Np, int K, int Nstore, float alpha, int relu)
{
  const int wave = threadIdx.x >> 5, lane = threadIdx.x & 31;
  const int wm = wave & 3, wn = wave >> 2;
  const int tm = blockIdx.x*128 + wm*32;
  const int tn = blockIdx.y*64  + wn*32;
  if (tm >= M || tn >= Np) return;                 // wave-uniform exit
  const int r16 = lane & 15, kh = lane >> 4;
  const half_t* Ar0 = A + (size_t)(tm + r16)*lda;
  const half_t* Ar1 = A + (size_t)(tm + 16 + r16)*lda;
  const half_t* Br0 = B + (size_t)(tn + r16)*ldb;
  const half_t* Br1 = B + (size_t)(tn + 16 + r16)*ldb;
  v8f acc00 = {}, acc01 = {}, acc10 = {}, acc11 = {};
  for (int k = 0; k < K; k += 32) {
    __builtin_prefetch(Ar0 + k + 256, 0, 3);
    __builtin_prefetch(Br0 + k + 256, 0, 3);
    v16h a0 = ldA(Ar0, k, kh);
    v16h a1 = ldA(Ar1, k, kh);
    v16h b0 = ldB(Br0, k, kh);
    v16h b1 = ldB(Br1, k, kh);
    acc00 = WMMA_(a0, b0, acc00);
    acc01 = WMMA_(a0, b1, acc01);
    acc10 = WMMA_(a1, b0, acc10);
    acc11 = WMMA_(a1, b1, acc11);
  }
  const int cn0 = tn + r16, cn1 = tn + 16 + r16;
  const int cm0 = tm + kh*8, cm1 = tm + 16 + kh*8;
#define STORE_(ACC, CM, CN)                                                  \
  if ((CN) < Nstore) {                                                       \
    float bv_ = bias ? bias[CN] : 0.f;                                       \
    _Pragma("unroll")                                                        \
    for (int r = 0; r < 8; ++r) {                                            \
      float v = ACC[r]*alpha + bv_;                                          \
      if (res)  v += res[(size_t)((CM)+r)*ldres + (CN)];                     \
      if (relu && v < 0.f) v = 0.f;                                          \
      C[(size_t)((CM)+r)*ldc + (CN)] = v;                                    \
    }                                                                        \
  }
  STORE_(acc00, cm0, cn0)
  STORE_(acc01, cm0, cn1)
  STORE_(acc10, cm1, cn0)
  STORE_(acc11, cm1, cn1)
#undef STORE_
}

// ------------------------------ staging kernels ----------------------------
__global__ void k_pack_wt(const float* __restrict__ w, int din, int dout,
                          half_t* __restrict__ wt, int Kp, int Np)
{
  int i = blockIdx.x*256 + threadIdx.x;
  if (i >= Np*Kp) return;
  int n = i / Kp, k = i % Kp;
  float v = (n < dout && k < din) ? w[(size_t)k*dout + n] : 0.f;
  wt[i] = (half_t)v;
}

__global__ void k_cast_pad(const float* __restrict__ src, int ldsrc, int M, int Klog,
                           half_t* __restrict__ dst, int Kp)
{
  int idx = blockIdx.x*256 + threadIdx.x;
  if (idx >= M*Kp) return;
  int row = idx / Kp, col = idx % Kp;
  float v = (col < Klog) ? src[(size_t)row*ldsrc + col] : 0.f;
  dst[idx] = (half_t)v;
}

__global__ void k_castT_pad(const float* __restrict__ src, int ldsrc, int R, int C,
                            half_t* __restrict__ dst, int Rp)
{
  int idx = blockIdx.x*256 + threadIdx.x;
  if (idx >= C*Rp) return;
  int c = idx / Rp, r = idx % Rp;
  float v = (r < R) ? src[(size_t)r*ldsrc + c] : 0.f;
  dst[idx] = (half_t)v;
}

// ----------------------------- elementwise etc -----------------------------
__global__ void k_copy(float* __restrict__ d, const float* __restrict__ s, int n){
  int i = blockIdx.x*256 + threadIdx.x; if (i < n) d[i] = s[i];
}
__global__ void k_copy2d(float* __restrict__ d, int ldd, const float* __restrict__ s,
                         int lds_, int M, int D){
  int i = blockIdx.x*256 + threadIdx.x; if (i >= M*D) return;
  int r = i / D, c = i % D; d[(size_t)r*ldd + c] = s[(size_t)r*lds_ + c];
}
__global__ void k_add(const float* __restrict__ a, const float* __restrict__ b,
                      float* __restrict__ o, int n){
  int i = blockIdx.x*256 + threadIdx.x; if (i < n) o[i] = a[i] + b[i];
}
// dst = src * m1[row] * (m2 ? (1-m2[row]) : 1)
__global__ void k_rowscale(const float* __restrict__ src, const float* __restrict__ m1,
                           const float* __restrict__ m2, float* __restrict__ dst,
                           int M, int D){
  int i = blockIdx.x*256 + threadIdx.x; if (i >= M*D) return;
  int r = i / D;
  float s = (m1 ? m1[r] : 1.f) * (m2 ? (1.f - m2[r]) : 1.f);
  dst[i] = src[i] * s;
}
__global__ void k_concat(const float* __restrict__ a, int Da,
                         const float* __restrict__ b, int Db,
                         float* __restrict__ out, int M){
  int D = Da + Db;
  int idx = blockIdx.x*256 + threadIdx.x;
  if (idx >= M*D) return;
  int row = idx / D, col = idx % D;
  out[idx] = (col < Da) ? a[(size_t)row*Da + col] : b[(size_t)row*Db + col - Da];
}
__global__ void k_add_lat(const float* __restrict__ base, const float* __restrict__ delta,
                          float* __restrict__ out, int M, int Dl){
  int idx = blockIdx.x*256 + threadIdx.x;
  if (idx >= M*Dl) return;
  int row = idx / Dl, col = idx % Dl;
  out[idx] = base[idx] + delta[(size_t)row*16 + col];
}
__global__ void k_padbias(const float* __restrict__ mask, float* __restrict__ pb){
  int j = threadIdx.x; if (j < N_) pb[j] = -INFB * (1.f - mask[j]);
}

// --------------------------------- layernorm -------------------------------
// out = LN(x + addmask[row]*add) * g + b, then * postmask[row]. In-place safe.
__global__ __launch_bounds__(256)
void k_ln(const float* __restrict__ x, const float* __restrict__ add,
          const float* __restrict__ addmask, const float* __restrict__ g,
          const float* __restrict__ b, float* __restrict__ out,
          int D, const float* __restrict__ postmask)
{
  int row = blockIdx.x, t = threadIdx.x;
  __shared__ float red[256];
  const float* xr = x + (size_t)row*D;
  const float* ar = add ? add + (size_t)row*D : nullptr;
  float am = addmask ? addmask[row] : 1.f;
  float s = 0.f;
  for (int c = t; c < D; c += 256) { float v = xr[c] + (ar ? ar[c]*am : 0.f); s += v; }
  red[t] = s; __syncthreads();
  for (int o = 128; o > 0; o >>= 1){ if (t < o) red[t] += red[t+o]; __syncthreads(); }
  float mean = red[0] / D; __syncthreads();
  float s2 = 0.f;
  for (int c = t; c < D; c += 256){
    float v = xr[c] + (ar ? ar[c]*am : 0.f) - mean; s2 += v*v;
  }
  red[t] = s2; __syncthreads();
  for (int o = 128; o > 0; o >>= 1){ if (t < o) red[t] += red[t+o]; __syncthreads(); }
  float inv = rsqrtf(red[0]/D + 1e-5f);
  float pm = postmask ? postmask[row] : 1.f;
  __syncthreads();
  for (int c = t; c < D; c += 256){
    float v = xr[c] + (ar ? ar[c]*am : 0.f);
    out[(size_t)row*D + c] = ((v - mean)*inv*g[c] + b[c]) * pm;
  }
}

// --------------------------------- softmax ---------------------------------
__global__ __launch_bounds__(256)
void k_softmax(const float* __restrict__ lg, const float* __restrict__ colbias,
               float* __restrict__ out, half_t* __restrict__ out16)
{
  int row = blockIdx.x, t = threadIdx.x;
  __shared__ float red[256];
  float v = lg[(size_t)row*N_ + t] + (colbias ? colbias[t] : 0.f);
  red[t] = v; __syncthreads();
  for (int o = 128; o > 0; o >>= 1){ if (t < o) red[t] = fmaxf(red[t], red[t+o]); __syncthreads(); }
  float mx = red[0]; __syncthreads();
  float e = expf(v - mx);
  red[t] = e; __syncthreads();
  for (int o = 128; o > 0; o >>= 1){ if (t < o) red[t] += red[t+o]; __syncthreads(); }
  float r = e / red[0];
  out[(size_t)row*N_ + t] = r;
  out16[(size_t)row*N_ + t] = (half_t)r;
}

// ------------------------------ IPA specifics ------------------------------
__global__ void k_initpose(const float* __restrict__ rig, float* __restrict__ quat,
                           float* __restrict__ trans){
  int n = threadIdx.x; if (n >= N_) return;
  for (int c = 0; c < 4; ++c) quat[n*4+c] = rig[n*7+c];
  for (int c = 0; c < 3; ++c) trans[n*3+c] = rig[n*7+4+c] * SCALE_;
}
__global__ void k_rot_pts(const float* __restrict__ quat, const float* __restrict__ trans,
                          const float* __restrict__ pts, float* __restrict__ outp, int HP){
  int idx = blockIdx.x*256 + threadIdx.x;
  if (idx >= N_*HP) return;
  int n = idx / HP;
  float R[9]; quatR(quat + n*4, R);
  const float* p = pts + (size_t)idx*3;
  float px = p[0], py = p[1], pz = p[2];
  outp[(size_t)idx*3+0] = R[0]*px + R[1]*py + R[2]*pz + trans[n*3+0];
  outp[(size_t)idx*3+1] = R[3]*px + R[4]*py + R[5]*pz + trans[n*3+1];
  outp[(size_t)idx*3+2] = R[6]*px + R[7]*py + R[8]*pz + trans[n*3+2];
}
// logits[h,i,j] += sqrt(1/3)*zb + pt_att + mask bias
__global__ void k_ipa_bias(float* __restrict__ logits, const float* __restrict__ zb,
                           const float* __restrict__ qpr, const float* __restrict__ kvr,
                           const float* __restrict__ head_w, const float* __restrict__ mask){
  int idx = blockIdx.x*256 + threadIdx.x;                  // H*N*N
  int j = idx & (N_-1); int i = (idx >> 8) & (N_-1); int h = idx >> 16;
  const float* qp = qpr + (((size_t)i*H_ + h)*PQ_)*3;
  const float* kp = kvr + (((size_t)j*H_ + h)*20)*3;
  float d2 = 0.f;
  for (int p = 0; p < PQ_; ++p){
    float dx = qp[p*3+0]-kp[p*3+0], dy = qp[p*3+1]-kp[p*3+1], dz = qp[p*3+2]-kp[p*3+2];
    d2 += dx*dx + dy*dy + dz*dz;
  }
  float w = head_w[h];
  float sp = (w > 20.f) ? w : log1pf(expf(w));             // softplus
  float hw = sp * sqrtf(1.f/(3.f*(PQ_*9.f/2.f)));
  float pt = -0.5f * hw * d2;
  float mb = INFB * (mask[i]*mask[j] - 1.f);
  logits[idx] += 0.57735026919f * zb[((size_t)i*N_ + j)*16 + h] + pt + mb;
}
// o_pt world coords: optw[i,h,p,c] = sum_j attn[h,i,j] * v_pts[j,h,p,c]
__global__ void k_opt(const float* __restrict__ attn, const float* __restrict__ kvr,
                      float* __restrict__ optw){
  int idx = blockIdx.x*256 + threadIdx.x;                  // N*H*PV*3
  if (idx >= N_*H_*PV_*3) return;
  int c = idx % 3; int t = idx/3; int p = t % PV_; t /= PV_; int h = t % H_; int i = t / H_;
  const float* ar = attn + ((size_t)h*N_ + i)*N_;
  float s = 0.f;
  for (int j = 0; j < N_; ++j)
    s += ar[j] * kvr[(((size_t)j*H_ + h)*20 + PQ_ + p)*3 + c];
  optw[idx] = s;
}
// inverse-rotate o_pt, write local coords + norms into cat
__global__ void k_optfin(const float* __restrict__ optw, const float* __restrict__ quat,
                         const float* __restrict__ trans, float* __restrict__ cat){
  int idx = blockIdx.x*256 + threadIdx.x;                  // N*H*PV
  if (idx >= N_*H_*PV_) return;
  int p = idx % PV_; int h = (idx/PV_) % H_; int i = idx/(PV_*H_);
  float R[9]; quatR(quat + i*4, R);
  const float* o = optw + (size_t)idx*3;
  float dx = o[0]-trans[i*3+0], dy = o[1]-trans[i*3+1], dz = o[2]-trans[i*3+2];
  float lx = R[0]*dx + R[3]*dy + R[6]*dz;   // R^T * d
  float ly = R[1]*dx + R[4]*dy + R[7]*dz;
  float lz = R[2]*dx + R[5]*dy + R[8]*dz;
  float* cr = cat + (size_t)i*3456;
  int hp = h*PV_ + p;
  cr[2048 + hp*3+0] = lx; cr[2048 + hp*3+1] = ly; cr[2048 + hp*3+2] = lz;
  cr[2336 + hp] = sqrtf(lx*lx + ly*ly + lz*lz + 1e-8f);
}
// o_pair: cat[i, 2432 + h*CZ + c] = sum_j attn[h,i,j]*edge[i,j,c]
__global__ void k_opair(const float* __restrict__ attn, const float* __restrict__ edge,
                        float* __restrict__ cat){
  int idx = blockIdx.x*256 + threadIdx.x;                  // N*H*CZ
  if (idx >= N_*H_*CZ_) return;
  int c = idx % CZ_; int h = (idx/CZ_) % H_; int i = idx/(CZ_*H_);
  const float* ar = attn + ((size_t)h*N_ + i)*N_;
  const float* er = edge + (size_t)i*N_*CZ_ + c;
  float s = 0.f;
  for (int j = 0; j < N_; ++j) s += ar[j] * er[(size_t)j*CZ_];
  cat[(size_t)i*3456 + 2432 + h*CZ_ + c] = s;
}
// backbone update: quat = normalize(quat * (1,u0..2)), trans += R_old * u3..5
__global__ void k_pose(float* __restrict__ quat, float* __restrict__ trans,
                       const float* __restrict__ upd, const float* __restrict__ mask,
                       const float* __restrict__ fixedm){
  int n = threadIdx.x; if (n >= N_) return;
  float dm = (1.f - fixedm[n]) * mask[n];
  float u[6];
  for (int c = 0; c < 6; ++c) u[c] = upd[n*16+c] * dm;
  float q0=quat[n*4], q1=quat[n*4+1], q2=quat[n*4+2], q3=quat[n*4+3];
  float R[9]; quatR(quat + n*4, R);
  float nw = q0       - q1*u[0] - q2*u[1] - q3*u[2];
  float nx = q0*u[0]  + q1      + q2*u[2] - q3*u[1];
  float ny = q0*u[1]  - q1*u[2] + q2      + q3*u[0];
  float nz = q0*u[2]  + q1*u[1] - q2*u[0] + q3;
  float inv = rsqrtf(nw*nw + nx*nx + ny*ny + nz*nz);
  quat[n*4]=nw*inv; quat[n*4+1]=nx*inv; quat[n*4+2]=ny*inv; quat[n*4+3]=nz*inv;
  for (int c = 0; c < 3; ++c)
    trans[n*3+c] += R[c*3+0]*u[3] + R[c*3+1]*u[4] + R[c*3+2]*u[5];
}
// edge-transition concat build: ee = [edge | ne_i | ne_j]
__global__ void k_edge_build(const float* __restrict__ edge, const float* __restrict__ ne,
                             float* __restrict__ ee, int row0, int CH){
  int idx = blockIdx.x*256 + threadIdx.x;
  if (idx >= CH*384) return;
  int col = idx % 384; int r = idx / 384; int pair = row0 + r;
  int i = pair >> 8, j = pair & (N_-1);
  float v;
  if (col < 128)      v = edge[(size_t)pair*128 + col];
  else if (col < 256) v = ne[i*128 + col - 128];
  else                v = ne[j*128 + col - 256];
  ee[idx] = v;
}
__global__ void k_emask(const float* __restrict__ mask, float* __restrict__ em,
                        int row0, int CH){
  int r = blockIdx.x*256 + threadIdx.x; if (r >= CH) return;
  int pair = row0 + r;
  em[r] = mask[pair >> 8] * mask[pair & (N_-1)];
}
__global__ void k_psi(const float* __restrict__ uns, float* __restrict__ out){
  int n = threadIdx.x; if (n >= N_) return;
  float a = uns[n*16], b = uns[n*16+1];
  float s = a*a + b*b; if (s < 1e-8f) s = 1e-8f;
  float inv = rsqrtf(s);
  out[n*2] = a*inv; out[n*2+1] = b*inv;
}
__global__ void k_rig7(const float* __restrict__ quat, const float* __restrict__ trans,
                       float* __restrict__ out){
  int n = threadIdx.x; if (n >= N_) return;
  for (int c = 0; c < 4; ++c) out[n*7+c] = quat[n*4+c];
  for (int c = 0; c < 3; ++c) out[n*7+4+c] = trans[n*3+c] / SCALE_;
}

// =============================== host driver ===============================
extern "C" void kernel_launch(void* const* d_in, const int* in_sizes, int n_in,
                              void* d_out, int out_size, void* d_ws, size_t ws_size,
                              hipStream_t stream) {
  (void)in_sizes; (void)n_in; (void)out_size; (void)ws_size;
  const float* in_node  = (const float*)d_in[0];
  const float* in_edge  = (const float*)d_in[1];
  const float* in_rig   = (const float*)d_in[2];
  const float* in_mask  = (const float*)d_in[3];
  const float* in_fixed = (const float*)d_in[4];
  const float* in_lat   = (const float*)d_in[5];
  const float* in_late  = (const float*)d_in[6];

  float* out      = (float*)d_out;
  float* out_psi  = out;              // [N,2]
  float* out_rig  = out + 512;        // [N,7]
  float* out_lat  = out + 2304;       // [N,4]
  float* out_late = out + 3328;       // [N,N,4]
  float* out_node = out + 265472;     // [N,256]

  // ---- workspace bump allocator (deterministic layout each call) ----
  char* wsB = (char*)d_ws; size_t off = 0;
  auto alloc = [&](size_t bytes)->void*{
    void* p = wsB + off; off = (off + bytes + 255) & ~(size_t)255; return p; };

  float*  node   = (float*) alloc((size_t)N_*256*4);
  float*  quat   = (float*) alloc((size_t)N_*4*4);
  float*  trans  = (float*) alloc((size_t)N_*3*4);
  float*  edge   = (float*) alloc((size_t)65536*128*4);
  half_t* z16    = (half_t*)alloc((size_t)65536*128*2);
  float*  qbuf   = (float*) alloc((size_t)N_*2048*4);
  float*  kvbuf  = (float*) alloc((size_t)N_*4096*4);
  float*  ptmp   = (float*) alloc((size_t)N_*480*4);
  float*  qpr    = (float*) alloc((size_t)N_*192*4);
  float*  kvpr   = (float*) alloc((size_t)N_*480*4);
  float*  zb     = (float*) alloc((size_t)65536*16*4);
  float*  logits = (float*) alloc((size_t)8*65536*4);
  float*  attn   = (float*) alloc((size_t)8*65536*4);
  half_t* a16    = (half_t*)alloc((size_t)8*65536*2);
  half_t* s16    = (half_t*)alloc((size_t)N_*3456*2);
  half_t* s16b   = (half_t*)alloc((size_t)N_*256*2);
  float*  cat    = (float*) alloc((size_t)N_*3456*4);
  float*  optw   = (float*) alloc((size_t)N_*H_*PV_*3*4);
  float*  ipaout = (float*) alloc((size_t)N_*256*4);
  float*  x      = (float*) alloc((size_t)N_*320*4);
  float*  qkvb   = (float*) alloc((size_t)N_*960*4);
  float*  obuf   = (float*) alloc((size_t)N_*320*4);
  float*  t1     = (float*) alloc((size_t)N_*1024*4);
  float*  t2     = (float*) alloc((size_t)N_*1024*4);
  float*  ne     = (float*) alloc((size_t)N_*128*4);
  float*  upd    = (float*) alloc((size_t)N_*16*4);
  float*  pb     = (float*) alloc((size_t)N_*4);
  float*  lf4    = (float*) alloc((size_t)N_*16*4);
  half_t* init16 = (half_t*)alloc((size_t)N_*256*2);
  half_t* w16    = (half_t*)alloc((size_t)4*1024*1024);     // packed weights
  const int CH = 8192;
  float*  ee  = (float*) alloc((size_t)CH*384*4);
  float*  eh  = (float*) alloc((size_t)CH*384*4);
  float*  et  = (float*) alloc((size_t)CH*384*4);
  half_t* e16 = (half_t*)alloc((size_t)CH*384*2);
  float*  ef  = (float*) alloc((size_t)CH*128*4);
  float*  em  = (float*) alloc((size_t)CH*4);

  // ---- param cursor: flattened pytree, alphabetical keys ----
  int cur = 7;
  auto nxt = [&]()->const float*{ return (const float*)d_in[cur++]; };
  struct Lin { const float* b; const float* w; int din, dout; };
  auto getlin = [&](int di, int dou)->Lin{ Lin l; l.b = nxt(); l.w = nxt();
                                           l.din = di; l.dout = dou; return l; };
  struct LNP { const float* b; const float* g; };
  auto getln = [&]()->LNP{ LNP l; l.b = nxt(); l.g = nxt(); return l; };

  // ---- launch helpers ----
  auto gemm = [&](const half_t* A16, int lda, const Lin& L, float* C, int ldc, int M,
                  float alpha, int relu, const float* res, int ldres){
    int Kp = pad32i(L.din), Np = pad16i(L.dout);
    int n = Np*Kp;
    k_pack_wt<<<dim3((n+255)/256), dim3(256), 0, stream>>>(L.w, L.din, L.dout, w16, Kp, Np);
    dim3 g((M+127)/128, (Np+63)/64);
    k_gemm<<<g, dim3(256), 0, stream>>>(A16, lda, w16, Kp, C, ldc, L.b,
                                        res, ldres, M, Np, Kp, L.dout, alpha, relu);
  };
  auto gemm_raw = [&](const half_t* A16, int lda, const half_t* B16, int ldb,
                      float* C, int ldc, int M, int Np, int K, int Nst, float alpha){
    dim3 g((M+127)/128, (Np+63)/64);
    k_gemm<<<g, dim3(256), 0, stream>>>(A16, lda, B16, ldb, C, ldc, nullptr,
                                        nullptr, 0, M, Np, K, Nst, alpha, 0);
  };
  auto cast = [&](const float* src, int ldsrc, int M, int K, half_t* dst, int Kp){
    int n = M*Kp;
    k_cast_pad<<<dim3((n+255)/256), dim3(256), 0, stream>>>(src, ldsrc, M, K, dst, Kp);
  };
  auto castT = [&](const float* src, int ldsrc, int R, int C2, half_t* dst, int Rp){
    int n = C2*Rp;
    k_castT_pad<<<dim3((n+255)/256), dim3(256), 0, stream>>>(src, ldsrc, R, C2, dst, Rp);
  };

  // ---- init ----
  k_initpose<<<1, 256, 0, stream>>>(in_rig, quat, trans);
  k_rowscale<<<dim3(256), dim3(256), 0, stream>>>(in_node, in_mask, nullptr, node, N_, 256);
  k_copy<<<dim3((65536*128+255)/256), dim3(256), 0, stream>>>(edge, in_edge, 65536*128);
  cast(node, 256, N_, 256, init16, 256);      // masked init_node (constant across blocks)
  k_padbias<<<1, 256, 0, stream>>>(in_mask, pb);

  const float qkscale = sqrtf(1.f/768.f);
  const float thscale = rsqrtf(80.f);

  for (int b = 0; b < 4; ++b) {
    // -------- parse block params (alphabetical) --------
    Lin bb = getlin(256, 6);
    bool hasE = (b < 3);
    Lin e_final{}, e_init{}, e_t0{}, e_t1{}; LNP e_ln{};
    if (hasE) { e_final = getlin(384,128); e_init = getlin(256,128);
                e_ln = getln(); e_t0 = getlin(384,384); e_t1 = getlin(384,384); }
    Lin ipa_b   = getlin(128, 8);
    const float* head_w = nxt();
    Lin ipa_kv  = getlin(256, 4096);
    Lin ipa_kvp = getlin(256, 480);
    Lin ipa_out = getlin(3456, 256);
    Lin ipa_q   = getlin(256, 2048);
    Lin ipa_qp  = getlin(256, 192);
    LNP ipa_ln  = getln();
    Lin post    = getlin(320, 256);
    Lin skip    = getlin(256, 64);
    struct TF { Lin ff1, ff2; LNP ln1, ln2; Lin o, qkv; } tf[2];
    for (int t = 0; t < 2; ++t) {
      tf[t].ff1 = getlin(320,320); tf[t].ff2 = getlin(320,320);
      tf[t].ln1 = getln(); tf[t].ln2 = getln();
      tf[t].o   = getlin(320,320); tf[t].qkv = getlin(320,960);
    }
    Lin t_l1 = getlin(256,256), t_l2 = getlin(256,256), t_l3 = getlin(256,256);
    LNP t_ln = getln();

    // -------- IPA --------
    cast(node, 256, N_, 256, s16, 256);
    gemm(s16, 256, ipa_q,   qbuf, 2048, N_, 1.f, 0, nullptr, 0);
    gemm(s16, 256, ipa_kv,  kvbuf,4096, N_, 1.f, 0, nullptr, 0);
    gemm(s16, 256, ipa_qp,  ptmp, 192,  N_, 1.f, 0, nullptr, 0);
    k_rot_pts<<<dim3((N_*64+255)/256), dim3(256), 0, stream>>>(quat, trans, ptmp, qpr, 64);
    gemm(s16, 256, ipa_kvp, ptmp, 480,  N_, 1.f, 0, nullptr, 0);
    k_rot_pts<<<dim3((N_*160+255)/256), dim3(256), 0, stream>>>(quat, trans, ptmp, kvpr, 160);
    cast(edge, 128, 65536, 128, z16, 128);
    gemm(z16, 128, ipa_b, zb, 16, 65536, 1.f, 0, nullptr, 0);
    for (int h = 0; h < H_; ++h) {
      cast(qbuf  + h*256, 2048, N_, 256, s16,  256);
      cast(kvbuf + h*512, 4096, N_, 256, s16b, 256);
      gemm_raw(s16, 256, s16b, 256, logits + (size_t)h*65536, 256,
               N_, 256, 256, 256, qkscale);
    }
    k_ipa_bias<<<dim3(8*65536/256), dim3(256), 0, stream>>>(logits, zb, qpr, kvpr,
                                                            head_w, in_mask);
    k_softmax<<<dim3(8*N_), dim3(256), 0, stream>>>(logits, nullptr, attn, a16);
    for (int h = 0; h < H_; ++h) {
      castT(kvbuf + h*512 + 256, 4096, N_, 256, s16b, 256);   // v^T
      gemm_raw(a16 + (size_t)h*65536, 256, s16b, 256, cat + h*256, 3456,
               N_, 256, 256, 256, 1.f);
    }
    k_opt   <<<dim3((N_*H_*PV_*3+255)/256), dim3(256), 0, stream>>>(attn, kvpr, optw);
    k_optfin<<<dim3((N_*H_*PV_+255)/256),   dim3(256), 0, stream>>>(optw, quat, trans, cat);
    k_opair <<<dim3(N_*H_*CZ_/256),         dim3(256), 0, stream>>>(attn, edge, cat);
    cast(cat, 3456, N_, 3456, s16, 3456);
    gemm(s16, 3456, ipa_out, ipaout, 256, N_, 1.f, 0, nullptr, 0);
    k_ln<<<dim3(N_), dim3(256), 0, stream>>>(node, ipaout, in_mask,
                                             ipa_ln.g, ipa_ln.b, node, 256, nullptr);

    // -------- transformer stack on [node | skip(init)] --------
    k_copy2d<<<dim3(256), dim3(256), 0, stream>>>(x, 320, node, 256, N_, 256);
    gemm(init16, 256, skip, x + 256, 320, N_, 1.f, 0, nullptr, 0);
    for (int t = 0; t < 2; ++t) {
      cast(x, 320, N_, 320, s16, 320);
      gemm(s16, 320, tf[t].qkv, qkvb, 960, N_, 1.f, 0, nullptr, 0);
      for (int h = 0; h < 4; ++h) {
        cast(qkvb + h*80,       960, N_, 80, s16,  96);
        cast(qkvb + 320 + h*80, 960, N_, 80, s16b, 96);
        gemm_raw(s16, 96, s16b, 96, logits, 256, N_, 256, 96, 256, thscale);
        k_softmax<<<dim3(N_), dim3(256), 0, stream>>>(logits, pb, attn, a16);
        castT(qkvb + 640 + h*80, 960, N_, 80, s16b, 256);
        gemm_raw(a16, 256, s16b, 256, obuf + h*80, 320, N_, 80, 256, 80, 1.f);
      }
      cast(obuf, 320, N_, 320, s16, 320);
      gemm(s16, 320, tf[t].o, t1, 320, N_, 1.f, 0, nullptr, 0);
      k_ln<<<dim3(N_), dim3(256), 0, stream>>>(x, t1, nullptr, tf[t].ln1.g,
                                               tf[t].ln1.b, x, 320, nullptr);
      cast(x, 320, N_, 320, s16, 320);
      gemm(s16, 320, tf[t].ff1, t1, 320, N_, 1.f, 1, nullptr, 0);
      cast(t1, 320, N_, 320, s16, 320);
      gemm(s16, 320, tf[t].ff2, t2, 320, N_, 1.f, 0, nullptr, 0);
      k_ln<<<dim3(N_), dim3(256), 0, stream>>>(x, t2, nullptr, tf[t].ln2.g,
                                               tf[t].ln2.b, x, 320, nullptr);
    }
    cast(x, 320, N_, 320, s16, 320);
    gemm(s16, 320, post, node, 256, N_, 1.f, 0, node, 256);   // node += post(x)

    // -------- node transition MLP --------
    cast(node, 256, N_, 256, s16, 256);
    gemm(s16, 256, t_l1, t1, 256, N_, 1.f, 1, nullptr, 0);
    cast(t1, 256, N_, 256, s16, 256);
    gemm(s16, 256, t_l2, t2, 256, N_, 1.f, 1, nullptr, 0);
    cast(t2, 256, N_, 256, s16, 256);
    gemm(s16, 256, t_l3, t1, 256, N_, 1.f, 0, nullptr, 0);
    k_ln<<<dim3(N_), dim3(256), 0, stream>>>(node, t1, nullptr, t_ln.g, t_ln.b,
                                             node, 256, in_mask);

    // -------- backbone update --------
    k_rowscale<<<dim3(256), dim3(256), 0, stream>>>(node, in_mask, in_fixed, t1, N_, 256);
    cast(t1, 256, N_, 256, s16, 256);
    gemm(s16, 256, bb, upd, 16, N_, 1.f, 0, nullptr, 0);
    k_pose<<<1, 256, 0, stream>>>(quat, trans, upd, in_mask, in_fixed);

    // -------- edge transition (chunked) --------
    if (hasE) {
      cast(node, 256, N_, 256, s16, 256);
      gemm(s16, 256, e_init, ne, 128, N_, 1.f, 0, nullptr, 0);
      for (int r0 = 0; r0 < 65536; r0 += CH) {
        k_edge_build<<<dim3(CH*384/256), dim3(256), 0, stream>>>(edge, ne, ee, r0, CH);
        cast(ee, 384, CH, 384, e16, 384);
        gemm(e16, 384, e_t0, eh, 384, CH, 1.f, 1, nullptr, 0);
        cast(eh, 384, CH, 384, e16, 384);
        gemm(e16, 384, e_t1, et, 384, CH, 1.f, 1, nullptr, 0);
        k_add<<<dim3(CH*384/256), dim3(256), 0, stream>>>(et, ee, et, CH*384);
        cast(et, 384, CH, 384, e16, 384);
        gemm(e16, 384, e_final, ef, 128, CH, 1.f, 0, nullptr, 0);
        k_emask<<<dim3(CH/256), dim3(256), 0, stream>>>(in_mask, em, r0, CH);
        k_ln<<<dim3(CH), dim3(256), 0, stream>>>(ef, nullptr, nullptr, e_ln.g, e_ln.b,
                                                 edge + (size_t)r0*128, 128, em);
      }
    }
  }

  // -------- latent node head (ld) --------
  Lin ld0 = getlin(260,256), ld1 = getlin(256,256), ld2 = getlin(256,256),
      ldf = getlin(256,4); LNP ldln = getln();
  Lin le0 = getlin(132,128), le1 = getlin(128,128), le2 = getlin(128,128),
      lef = getlin(128,4); LNP leln = getln();
  Lin to1 = getlin(256,256), to2 = getlin(256,256), tof = getlin(256,2);

  k_concat<<<dim3((N_*260+255)/256), dim3(256), 0, stream>>>(in_lat, 4, node, 256, t1, N_);
  cast(t1, 260, N_, 260, s16, 288);
  gemm(s16, 288, ld0, t2, 256, N_, 1.f, 1, nullptr, 0);
  cast(t2, 256, N_, 256, s16, 256);
  gemm(s16, 256, ld1, t1, 256, N_, 1.f, 1, nullptr, 0);
  cast(t1, 256, N_, 256, s16, 256);
  gemm(s16, 256, ld2, t2, 256, N_, 1.f, 0, nullptr, 0);
  k_ln<<<dim3(N_), dim3(256), 0, stream>>>(t2, nullptr, nullptr, ldln.g, ldln.b,
                                           t1, 256, nullptr);
  cast(t1, 256, N_, 256, s16, 256);
  gemm(s16, 256, ldf, lf4, 16, N_, 1.f, 0, nullptr, 0);
  k_add_lat<<<dim3((N_*4+255)/256), dim3(256), 0, stream>>>(in_lat, lf4, out_lat, N_, 4);

  // -------- latent edge head (led), chunked --------
  for (int r0 = 0; r0 < 65536; r0 += CH) {
    k_concat<<<dim3((CH*132+255)/256), dim3(256), 0, stream>>>(
        in_late + (size_t)r0*4, 4, edge + (size_t)r0*128, 128, ee, CH);
    cast(ee, 132, CH, 132, e16, 160);
    gemm(e16, 160, le0, eh, 128, CH, 1.f, 1, nullptr, 0);
    cast(eh, 128, CH, 128, e16, 128);
    gemm(e16, 128, le1, et, 128, CH, 1.f, 1, nullptr, 0);
    cast(et, 128, CH, 128, e16, 128);
    gemm(e16, 128, le2, eh, 128, CH, 1.f, 0, nullptr, 0);
    k_ln<<<dim3(CH), dim3(256), 0, stream>>>(eh, nullptr, nullptr, leln.g, leln.b,
                                             et, 128, nullptr);
    cast(et, 128, CH, 128, e16, 128);
    gemm(e16, 128, lef, ef, 16, CH, 1.f, 0, nullptr, 0);
    k_add_lat<<<dim3(CH*4/256), dim3(256), 0, stream>>>(
        in_late + (size_t)r0*4, ef, out_late + (size_t)r0*4, CH, 4);
  }

  // -------- torsion head + outputs --------
  cast(node, 256, N_, 256, s16, 256);
  gemm(s16, 256, to1, t1, 256, N_, 1.f, 1, nullptr, 0);
  cast(t1, 256, N_, 256, s16, 256);
  gemm(s16, 256, to2, t2, 256, N_, 1.f, 0, node, 256);        // s = l2(relu(l1)) + node
  cast(t2, 256, N_, 256, s16, 256);
  gemm(s16, 256, tof, lf4, 16, N_, 1.f, 0, nullptr, 0);
  k_psi <<<1, 256, 0, stream>>>(lf4, out_psi);
  k_rig7<<<1, 256, 0, stream>>>(quat, trans, out_rig);
  k_copy<<<dim3(256), dim3(256), 0, stream>>>(out_node, node, N_*256);
}
